// SpikLinearMaxLayer_24283745092265
// MI455X (gfx1250) — matrix operand order repaired and checked by reading, not compiled, not run on hardware
//
#include <hip/hip_runtime.h>
#include <stdint.h>

// ---------------------------------------------------------------------------
// Fused SpikingJelly-style layer for MI455X (gfx1250, wave32, WMMA):
//   h = x @ W^T  (bf16 WMMA, fp32 accum)  -> BN affine -> maxpool2 -> LIF scan
// x: [T=4, B*N=32768, D=512] fp32, W: [O=512, D=512] fp32
// out spikes: [T, B*N, O/2=256] fp32
//
// Loop order: t-pair -> panel -> kk -> (t in pair) -> tile.
// B fragments (LDS) load once per kk and feed 8 WMMAs (2 t x 4 tiles).
// acc[2][4] = 64 VGPRs keeps total pressure < 256 (no scratch spills).
// LIF membrane state persists in registers across the two t-passes.
// ---------------------------------------------------------------------------

typedef __attribute__((ext_vector_type(16))) __bf16 v16bf;
typedef __attribute__((ext_vector_type(16))) float  v16f;
typedef __attribute__((ext_vector_type(8)))  float  v8f;
typedef __attribute__((ext_vector_type(4)))  float  v4f;
typedef __attribute__((ext_vector_type(4)))  __bf16 v4bf;

constexpr int kT  = 4;
constexpr int kTB = 2;         // time steps per pass (accumulator blocking)
constexpr int kBN = 32768;     // B*N
constexpr int kD  = 512;
constexpr int kO  = 512;
constexpr int kPO = 256;       // pooled channels
constexpr float kEps = 1e-5f;

constexpr int M_TILE  = 128;   // 8 waves * 16 rows
constexpr int O_TILE  = 64;    // 4 column tiles of 16
constexpr int K_PANEL = 256;   // W staged in 2 bf16 panels in LDS
constexpr int LDS_STR = K_PANEL + 8;  // pad: row stride 528B -> conflict-free b128

union FragB16 {
  v16bf v;
  uint4 q[2];
};

union Pack4 {
  v4bf v;
  uint2 q;
};

__global__ __launch_bounds__(256)
void snn_linear_bn_pool_lif(const float* __restrict__ x,
                            const float* __restrict__ Wm,
                            const float* __restrict__ gamma,
                            const float* __restrict__ beta,
                            const float* __restrict__ rmean,
                            const float* __restrict__ rvar,
                            float* __restrict__ out) {
  __shared__ unsigned short Wlds[O_TILE][LDS_STR];   // ~33 KB bf16 W panel

  const int tid   = threadIdx.x;
  const int wave  = tid >> 5;
  const int lane  = tid & 31;
  const int col   = lane & 15;      // WMMA column / A-row slot within wave tile
  const int khalf = lane >> 4;      // K-half selector per ISA 16-bit layouts

  const int ocBase  = blockIdx.x * O_TILE;                // fast dim -> x reuse in L2
  const int rowBase = blockIdx.y * M_TILE + wave * 16;
  const int arow    = rowBase + col;  // row this lane feeds into A fragments

  // BN affine constants for this lane's 4 channel columns
  float inv[4], shf[4];
#pragma unroll
  for (int tile = 0; tile < 4; ++tile) {
    int o = ocBase + tile * 16 + col;
    float iv = gamma[o] * rsqrtf(rvar[o] + kEps);
    inv[tile] = iv;
    shf[tile] = beta[o] - rmean[o] * iv;
  }

  // LIF membrane state (pooled); persists across both t-passes.
  float v[4][8];
#pragma unroll
  for (int tile = 0; tile < 4; ++tile)
#pragma unroll
    for (int j = 0; j < 8; ++j) v[tile][j] = 0.f;

  for (int tg = 0; tg < kT / kTB; ++tg) {
    // Accumulators for this pair of time steps.
    v8f acc[kTB][4] = {};

    for (int p = 0; p < 2; ++p) {
      const int kp = p * K_PANEL;

      __syncthreads();   // previous panel fully consumed
      // Cooperative stage: W[ocBase..+63, kp..kp+255] fp32 -> bf16 LDS
      for (int idx = tid; idx < O_TILE * (K_PANEL / 4); idx += 256) {
        const int c  = idx >> 6;              // 64 float4 chunks per column
        const int dd = (idx & 63) << 2;
        const float4 w4 = *(const float4*)&Wm[(size_t)(ocBase + c) * kD + kp + dd];
        v4f wf; wf[0] = w4.x; wf[1] = w4.y; wf[2] = w4.z; wf[3] = w4.w;
        Pack4 pk;
        pk.v = __builtin_convertvector(wf, v4bf);
        *(uint2*)&Wlds[c][dd] = pk.q;
      }
      __syncthreads();

#pragma unroll 4
      for (int kk = 0; kk < K_PANEL; kk += 32) {
        // Load all 4 B fragments once; they feed 8 WMMAs below.
        FragB16 bf_[4];
#pragma unroll
        for (int tile = 0; tile < 4; ++tile) {
          const uint4* src = (const uint4*)&Wlds[tile * 16 + col][kk + khalf * 16];
          bf_[tile].q[0] = src[0];
          bf_[tile].q[1] = src[1];
        }

#pragma unroll
        for (int tt = 0; tt < kTB; ++tt) {
          const int t = tg * kTB + tt;
          // A fragment, contiguous K order (matches B):
          //   lanes<16: K in {0..15}; lanes>=16: K in {16..31}
          const float* xrow =
              x + ((size_t)t * kBN + arow) * kD + kp + kk + khalf * 16;
          const float4 a0 = *(const float4*)&xrow[0];
          const float4 a1 = *(const float4*)&xrow[4];
          const float4 a2 = *(const float4*)&xrow[8];
          const float4 a3 = *(const float4*)&xrow[12];
          v16f afl;
          afl[0]  = a0.x; afl[1]  = a0.y; afl[2]  = a0.z; afl[3]  = a0.w;
          afl[4]  = a1.x; afl[5]  = a1.y; afl[6]  = a1.z; afl[7]  = a1.w;
          afl[8]  = a2.x; afl[9]  = a2.y; afl[10] = a2.z; afl[11] = a2.w;
          afl[12] = a3.x; afl[13] = a3.y; afl[14] = a3.z; afl[15] = a3.w;
          const v16bf av = __builtin_convertvector(afl, v16bf);

#pragma unroll
          for (int tile = 0; tile < 4; ++tile) {
            acc[tt][tile] = __builtin_amdgcn_wmma_f32_16x16x32_bf16(
                false, av, false, bf_[tile].v, (short)0, acc[tt][tile],
                false, false);
          }
        }
      }
    }

    // ---- Epilogue for this t-pair: BN affine -> max-pool -> LIF step ----
#pragma unroll
    for (int tt = 0; tt < kTB; ++tt) {
      const int t = tg * kTB + tt;
#pragma unroll
      for (int tile = 0; tile < 4; ++tile) {
        const int o = ocBase + tile * 16 + col;
        const size_t obase =
            ((size_t)t * kBN + rowBase + khalf * 8) * kPO + (size_t)(o >> 1);
#pragma unroll
        for (int j = 0; j < 8; ++j) {
          // C layout: lanes 0-15 hold M=j, lanes 16-31 hold M=j+8, N=col
          float h  = acc[tt][tile][j] * inv[tile] + shf[tile];
          float hp = fmaxf(h, __shfl_xor(h, 1, 32));    // pool channels (2p,2p+1)
          float vv = 0.5f * (v[tile][j] + hp);          // v += (h - v)/tau, tau=2
          float s  = (vv >= 1.0f) ? 1.0f : 0.0f;        // Heaviside(v - vth)
          v[tile][j] = vv * (1.0f - s);                 // hard reset (detached)
          if ((col & 1) == 0) {
            out[obase + (size_t)j * kPO] = s;
          }
        }
      }
    }
  }
}

extern "C" void kernel_launch(void* const* d_in, const int* in_sizes, int n_in,
                              void* d_out, int out_size, void* d_ws, size_t ws_size,
                              hipStream_t stream) {
  (void)in_sizes; (void)n_in; (void)d_ws; (void)ws_size; (void)out_size;
  const float* x     = (const float*)d_in[0];
  const float* Wm    = (const float*)d_in[1];
  const float* gamma = (const float*)d_in[2];
  const float* beta  = (const float*)d_in[3];
  const float* rmean = (const float*)d_in[4];
  const float* rvar  = (const float*)d_in[5];
  float* out = (float*)d_out;

  dim3 grid(kO / O_TILE, kBN / M_TILE);   // (8, 256): O-tile fastest -> x L2 reuse
  snn_linear_bn_pool_lif<<<grid, 256, 0, stream>>>(x, Wm, gamma, beta,
                                                   rmean, rvar, out);
}
